// ReadoutFunction_9655086481680
// MI455X (gfx1250) — matrix operand order repaired and checked by reading, not compiled, not run on hardware
//
#include <hip/hip_runtime.h>
#include <hip/hip_bf16.h>

// ---------------------------------------------------------------------------
// MI455X (gfx1250) readout kernel.
//
// Shapes: h0,hT [512,256,256] f32.  gate MLP 512->128->256->128->256 (ReLU,
// sigmoid at end), value MLP 256->128->256->128->256 (ReLU, linear end),
// out[b,t] = sum_n mask(b,n) * gate(b,n,t) * val(b,n,t).
//
// Strategy: bf16 WMMA (v_wmma_f32_16x16x32_bf16) for every matmul.
//  - Weights pre-swizzled to B-fragment order in d_ws (bf16).
//  - 16-row tiles per wave; layer-0 A fragments straight from global f32
//    (single HBM pass over the 268 MB of activations), later layers via
//    wave-private LDS ping-pong buffers.
// ---------------------------------------------------------------------------

typedef __attribute__((ext_vector_type(4)))  float  f32x4;
typedef __attribute__((ext_vector_type(8)))  float  v8f;
typedef __attribute__((ext_vector_type(16))) __bf16 v16bf;

union FragBF { f32x4 f4[2]; v16bf v; };

__device__ __forceinline__ void wave_lds_fence() {
  // Wave-private LDS RAW across lanes: LDS ops are in-order per wave; the
  // barrier+wait just pins compiler ordering and drains DScnt.
  __builtin_amdgcn_wave_barrier();
  asm volatile("s_wait_dscnt 0" ::: "memory");
  __builtin_amdgcn_wave_barrier();
}

// B fragment (K=32 x N=16 bf16), pre-swizzled: lane l holds 16 contiguous
// bf16 = column n=(l&15), k = kt*32 + (j>>3)*16 + (l>>4)*8 + (j&7)  (ISA 7.12.2 B layout).
__device__ __forceinline__ v16bf load_b_frag(const __bf16* __restrict__ W, int f, int lane) {
  FragBF fb;
  const f32x4* p = (const f32x4*)(W + f * 512 + lane * 16);
  fb.f4[0] = p[0];
  fb.f4[1] = p[1];
  return fb.v;
}

// A fragment (M=16 x K=32 bf16) from a row-major [16 x K] bf16 LDS tile.
// lane: row m=lane&15; low lanes K {0..7,16..23}, high lanes K {8..15,24..31}.
__device__ __forceinline__ v16bf load_a_lds(const __bf16* act, int K, int kt, int lane) {
  int m = lane & 15, h = lane >> 4;
  const __bf16* row = act + m * K + kt * 32 + h * 8;
  FragBF fa;
  fa.f4[0] = *(const f32x4*)(row);
  fa.f4[1] = *(const f32x4*)(row + 16);
  return fa.v;
}

// A fragment straight from global f32 activations (row-major, ld=256),
// converting f32 -> bf16 in registers.
__device__ __forceinline__ v16bf load_a_global_f32(const float* __restrict__ src,
                                                   int row0, int k0, int lane) {
  int m = lane & 15, h = lane >> 4;
  const float* p = src + (size_t)(row0 + m) * 256 + k0 + h * 8;
  f32x4 x0 = *(const f32x4*)(p);
  f32x4 x1 = *(const f32x4*)(p + 4);
  f32x4 y0 = *(const f32x4*)(p + 16);
  f32x4 y1 = *(const f32x4*)(p + 20);
  v16bf a;
#pragma unroll
  for (int j = 0; j < 4; ++j) {
    a[j]      = (__bf16)x0[j];
    a[4 + j]  = (__bf16)x1[j];
    a[8 + j]  = (__bf16)y0[j];
    a[12 + j] = (__bf16)y1[j];
  }
  return a;
}

// C/D layout: VGPR v -> row (lane>=16 ? v+8 : v), col = lane&15 within n-tile.
__device__ __forceinline__ v8f bias_acc(const float* __restrict__ bias, int nt, int lane) {
  float bn = bias[nt * 16 + (lane & 15)];
  v8f acc;
#pragma unroll
  for (int i = 0; i < 8; ++i) acc[i] = bn;
  return acc;
}

#define WMMA_BF16(a, b, c) \
  __builtin_amdgcn_wmma_f32_16x16x32_bf16(false, (a), false, (b), (short)0, (c), false, false)

// Generic hidden layer: LDS bf16 [16 x K] -> LDS bf16 [16 x NOUT].
template<int K, int NOUT, bool RELU>
__device__ __forceinline__ void layer_lds(const __bf16* actin, const __bf16* __restrict__ W,
                                          const float* __restrict__ bias, __bf16* actout, int lane) {
  int h = lane >> 4, nlo = lane & 15;
  for (int nt = 0; nt < NOUT / 16; ++nt) {
    v8f acc = bias_acc(bias, nt, lane);
#pragma unroll
    for (int kt = 0; kt < K / 32; ++kt) {
      v16bf a = load_a_lds(actin, K, kt, lane);
      v16bf b = load_b_frag(W, kt * (NOUT / 16) + nt, lane);
      acc = WMMA_BF16(a, b, acc);
    }
    int n = nt * 16 + nlo;
#pragma unroll
    for (int v = 0; v < 8; ++v) {
      float x = acc[v];
      if (RELU) x = fmaxf(x, 0.0f);
      actout[(h * 8 + v) * NOUT + n] = (__bf16)x;
    }
  }
}

// Layer 0: A comes from global f32 (optionally concat(h0,hT) along K).
template<int NOUT, bool CONCAT>
__device__ __forceinline__ void layer0_global(const float* __restrict__ h0,
                                              const float* __restrict__ hT,
                                              int row0, const __bf16* __restrict__ W,
                                              const float* __restrict__ bias,
                                              __bf16* actout, int lane) {
  const int KT = CONCAT ? 16 : 8;  // k-tiles of 32 (K = 512 or 256)
  int h = lane >> 4, nlo = lane & 15;
  for (int nt = 0; nt < NOUT / 16; ++nt) {
    v8f acc = bias_acc(bias, nt, lane);
#pragma unroll
    for (int kt = 0; kt < KT; ++kt) {
      const float* src = (CONCAT && (kt >= 8)) ? hT : (CONCAT ? h0 : hT);
      v16bf a = load_a_global_f32(src, row0, (kt & 7) * 32, lane);
      v16bf b = load_b_frag(W, kt * (NOUT / 16) + nt, lane);
      acc = WMMA_BF16(a, b, acc);
    }
    int n = nt * 16 + nlo;
#pragma unroll
    for (int v = 0; v < 8; ++v)
      actout[(h * 8 + v) * NOUT + n] = (__bf16)fmaxf(acc[v], 0.0f);
  }
}

// Value final layer: linear, f32 LDS output [16 x NOUT].
template<int K, int NOUT>
__device__ __forceinline__ void layer_lds_to_f32(const __bf16* actin, const __bf16* __restrict__ W,
                                                 const float* __restrict__ bias, float* outv, int lane) {
  int h = lane >> 4, nlo = lane & 15;
  for (int nt = 0; nt < NOUT / 16; ++nt) {
    v8f acc = bias_acc(bias, nt, lane);
#pragma unroll
    for (int kt = 0; kt < K / 32; ++kt) {
      v16bf a = load_a_lds(actin, K, kt, lane);
      v16bf b = load_b_frag(W, kt * (NOUT / 16) + nt, lane);
      acc = WMMA_BF16(a, b, acc);
    }
    int n = nt * 16 + nlo;
#pragma unroll
    for (int v = 0; v < 8; ++v)
      outv[(h * 8 + v) * NOUT + n] = acc[v];
  }
}

// Gate final layer fused with sigmoid * mask * val, row-reduction, atomicAdd.
__device__ __forceinline__ void gate_final(const __bf16* actin /*16x128*/,
                                           const __bf16* __restrict__ W,
                                           const float* __restrict__ bias,
                                           const float* bufV /*16x256 f32*/,
                                           const float* maskv /*8 rows per lane-half*/,
                                           float* __restrict__ outrow, int lane) {
  int h = lane >> 4, nlo = lane & 15;
  for (int nt = 0; nt < 16; ++nt) {
    v8f acc = bias_acc(bias, nt, lane);
#pragma unroll
    for (int kt = 0; kt < 4; ++kt) {
      v16bf a = load_a_lds(actin, 128, kt, lane);
      v16bf b = load_b_frag(W, kt * 16 + nt, lane);
      acc = WMMA_BF16(a, b, acc);
    }
    float s = 0.0f;
#pragma unroll
    for (int v = 0; v < 8; ++v) {
      float g   = 1.0f / (1.0f + __expf(-acc[v]));
      float val = bufV[(h * 8 + v) * 256 + nt * 16 + nlo];
      s += maskv[v] * g * val;
    }
    s += __shfl_xor(s, 16, 32);       // combine the two 8-row halves
    if (lane < 16) atomicAdd(outrow + nt * 16 + nlo, s);
  }
}

// ---------------------------------------------------------------------------
// Main kernel: 4 waves/block, one 16-row tile per wave.
// Dynamic LDS per wave: bufA(8KB bf16) | bufB(8KB bf16) | bufV(16KB f32) = 32KB
// + mask area; 128KB+256B per block (CDNA5 allows 320KB/WGP).
// ---------------------------------------------------------------------------
__global__ __launch_bounds__(128) void readout_kernel(
    const float* __restrict__ h0, const float* __restrict__ hT,
    const __bf16* __restrict__ gw0, const __bf16* __restrict__ gw1,
    const __bf16* __restrict__ gw2, const __bf16* __restrict__ gw3,
    const __bf16* __restrict__ ow0, const __bf16* __restrict__ ow1,
    const __bf16* __restrict__ ow2, const __bf16* __restrict__ ow3,
    const float* __restrict__ gb0, const float* __restrict__ gb1,
    const float* __restrict__ gb2, const float* __restrict__ gb3,
    const float* __restrict__ ob0, const float* __restrict__ ob1,
    const float* __restrict__ ob2, const float* __restrict__ ob3,
    float* __restrict__ out) {
  extern __shared__ char smem[];
  const int lane = threadIdx.x & 31;
  const int wave = threadIdx.x >> 5;
  const int tile = blockIdx.x * 4 + wave;  // 8192 tiles of 16 rows, exact cover
  const int row0 = tile * 16;
  const int b    = row0 >> 8;              // 256 nodes per batch entry

  char*   wbase = smem + wave * 32768;
  __bf16* bufA  = (__bf16*)(wbase);
  __bf16* bufB  = (__bf16*)(wbase + 8192);
  float*  bufV  = (float*)(wbase + 16384);
  float*  maskb = (float*)(smem + 4 * 32768) + wave * 16;

  // Warm the weight path (L2-resident after first touch): global_prefetch_b8.
  __builtin_prefetch(gw0 + lane * 512, 0, 1);
  __builtin_prefetch(ow0 + lane * 512, 0, 1);

  // ---- node-validity mask: sum(h0 row) > 0 -------------------------------
  {
    int m = lane & 15, h = lane >> 4;
    const f32x4* p = (const f32x4*)(h0 + (size_t)(row0 + m) * 256 + h * 128);
    float s = 0.0f;
#pragma unroll
    for (int i = 0; i < 32; ++i) {
      f32x4 x = p[i];
      s += x[0] + x[1] + x[2] + x[3];
    }
    s += __shfl_xor(s, 16, 32);
    if (lane < 16) maskb[m] = (s > 0.0f) ? 1.0f : 0.0f;
  }
  __syncthreads();
  float maskv[8];
  {
    int h = lane >> 4;
#pragma unroll
    for (int v = 0; v < 8; ++v) maskv[v] = maskb[h * 8 + v];
  }

  // ---- value MLP: hT -> bufV (f32) ---------------------------------------
  layer0_global<128, false>(hT, hT, row0, ow0, ob0, bufA, lane);
  wave_lds_fence();
  layer_lds<128, 256, true>(bufA, ow1, ob1, bufB, lane);
  wave_lds_fence();
  layer_lds<256, 128, true>(bufB, ow2, ob2, bufA, lane);
  wave_lds_fence();
  layer_lds_to_f32<128, 256>(bufA, ow3, ob3, bufV, lane);
  wave_lds_fence();

  // ---- gate MLP: concat(h0,hT) -> fused combine --------------------------
  layer0_global<128, true>(h0, hT, row0, gw0, gb0, bufA, lane);
  wave_lds_fence();
  layer_lds<128, 256, true>(bufA, gw1, gb1, bufB, lane);
  wave_lds_fence();
  layer_lds<256, 128, true>(bufB, gw2, gb2, bufA, lane);
  wave_lds_fence();
  gate_final(bufA, gw3, gb3, bufV, maskv, out + (size_t)b * 256, lane);
}

// ---------------------------------------------------------------------------
// Prepass: swizzle fp32 row-major [K,N] weights into bf16 B-fragment order.
// dest[f*512 + l*16 + j] = W[k*N + n], k = kt*32 + (j>>3)*16 + (l>>4)*8 + (j&7),
// n = nt*16 + (l&15), f = kt*(N/16) + nt.
// ---------------------------------------------------------------------------
__global__ void swizzle_weights(const float* __restrict__ src, __bf16* __restrict__ dst,
                                int K, int N, int total) {
  int id = blockIdx.x * blockDim.x + threadIdx.x;
  if (id >= total) return;
  int f = id >> 9;
  int r = id & 511;
  int l = r >> 4, j = r & 15;
  int ntiles = N >> 4;
  int kt = f / ntiles, nt = f % ntiles;
  int n = nt * 16 + (l & 15);
  int k = kt * 32 + (j >> 3) * 16 + (l >> 4) * 8 + (j & 7);
  dst[id] = (__bf16)src[k * N + n];
}

__global__ void zero_out_kernel(float* __restrict__ out, int n) {
  int i = blockIdx.x * blockDim.x + threadIdx.x;
  if (i < n) out[i] = 0.0f;
}

extern "C" void kernel_launch(void* const* d_in, const int* in_sizes, int n_in,
                              void* d_out, int out_size, void* d_ws, size_t ws_size,
                              hipStream_t stream) {
  (void)in_sizes; (void)n_in; (void)ws_size;
  const float* h0 = (const float*)d_in[0];
  const float* hT = (const float*)d_in[1];
  const float* gW[4] = {(const float*)d_in[2], (const float*)d_in[4],
                        (const float*)d_in[6], (const float*)d_in[8]};
  const float* gB[4] = {(const float*)d_in[3], (const float*)d_in[5],
                        (const float*)d_in[7], (const float*)d_in[9]};
  const float* oW[4] = {(const float*)d_in[10], (const float*)d_in[12],
                        (const float*)d_in[14], (const float*)d_in[16]};
  const float* oB[4] = {(const float*)d_in[11], (const float*)d_in[13],
                        (const float*)d_in[15], (const float*)d_in[17]};

  __bf16* ws = (__bf16*)d_ws;
  // gate: 512x128, 128x256, 256x128, 128x256 ; value: 256x128, 128x256, 256x128, 128x256
  const int gK[4] = {512, 128, 256, 128}, gN[4] = {128, 256, 128, 256};
  const int oK[4] = {256, 128, 256, 128}, oN[4] = {128, 256, 128, 256};
  size_t off = 0;
  __bf16* gws[4];
  __bf16* ows[4];
  for (int i = 0; i < 4; ++i) {
    gws[i] = ws + off;
    int total = gK[i] * gN[i];
    swizzle_weights<<<(total + 255) / 256, 256, 0, stream>>>(gW[i], gws[i], gK[i], gN[i], total);
    off += (size_t)total;
  }
  for (int i = 0; i < 4; ++i) {
    ows[i] = ws + off;
    int total = oK[i] * oN[i];
    swizzle_weights<<<(total + 255) / 256, 256, 0, stream>>>(oW[i], ows[i], oK[i], oN[i], total);
    off += (size_t)total;
  }

  zero_out_kernel<<<(out_size + 255) / 256, 256, 0, stream>>>((float*)d_out, out_size);

  const size_t smem_bytes = 4 * 32768 + 4 * 16 * sizeof(float);  // 128 KB + mask
  readout_kernel<<<dim3(2048), dim3(128), smem_bytes, stream>>>(
      h0, hT,
      gws[0], gws[1], gws[2], gws[3],
      ows[0], ows[1], ows[2], ows[3],
      gB[0], gB[1], gB[2], gB[3],
      oB[0], oB[1], oB[2], oB[3],
      (float*)d_out);
}
